// PerformerMultiHeadAttn_15891378995461
// MI455X (gfx1250) — matrix operand order repaired and checked by reading, not compiled, not run on hardware
//
#include <hip/hip_runtime.h>

// Performer (FAVOR+) multi-head attention, linear-time restructuring.
// L=2048, B=2, NH=8, DH=64, DM=512, M=256.
#define L_  2048
#define B_  2
#define NH_ 8
#define DH_ 64
#define DM_ 512
#define M_  256
#define N_  (L_*B_)      // 4096 token rows
#define RN_ (N_*NH_)     // 32768 (row, head) pairs

typedef __attribute__((ext_vector_type(16))) __bf16 v16bf;
typedef __attribute__((ext_vector_type(8)))  __bf16 v8bf;
typedef __attribute__((ext_vector_type(8)))  float  v8f;

static __device__ __forceinline__ __bf16 f2bf(float f) {
  unsigned u = __builtin_bit_cast(unsigned, f);
  unsigned short h = (unsigned short)((u + 0x7FFFu + ((u >> 16) & 1u)) >> 16);
  return __builtin_bit_cast(__bf16, h);
}
static __device__ __forceinline__ float bf2f(__bf16 b) {
  unsigned short hs = __builtin_bit_cast(unsigned short, b);
  unsigned u = ((unsigned)hs) << 16;
  return __builtin_bit_cast(float, u);
}

// WMMA 16x16x32 bf16 fragment, contiguous k order used here: per lane = two 8-element contiguous chunks.
// A (16x32): lanes 0-15 row M=lane hold K[0..15]; lanes 16-31 row
//            M=lane-16 hold K[16..31].
// B (32x16): lane's column holds 16 contiguous K (two 8-chunks at +0,+8).
static __device__ __forceinline__ v16bf frag(const __bf16* p0, const __bf16* p1) {
  v8bf a = *(const v8bf*)p0;
  v8bf b = *(const v8bf*)p1;
  return __builtin_shufflevector(a, b, 0,1,2,3,4,5,6,7,8,9,10,11,12,13,14,15);
}

#define WMMA_BF16(A, B, C) \
  __builtin_amdgcn_wmma_f32_16x16x32_bf16(false, (A), false, (B), (short)0, (C), false, false)

// 32x64 output tile per wave32: 2 A-row fragments x 4 B-col fragments ->
// 8 independent v_wmma per K-step (XDL ILP), f32 accumulate.
// a0r/a1r: &A[m0+(lane&15)][0], &A[m0+16+(lane&15)][0]
// b0r..b3r: &Bt[c0+16*j+(lane&15)][0]          half = lane>>4
static __device__ __forceinline__ void wave_gemm_2x4(
    const __bf16* __restrict__ a0r, const __bf16* __restrict__ a1r,
    const __bf16* __restrict__ b0r, const __bf16* __restrict__ b1r,
    const __bf16* __restrict__ b2r, const __bf16* __restrict__ b3r,
    int K, int half, v8f* acc) {
  for (int k0 = 0; k0 < K; k0 += 32) {
    int ka = k0 + half * 16;
    int kb = k0 + half * 16;
    if (k0 + 128 < K) {                       // global_prefetch_b8 ahead
      __builtin_prefetch(a0r + k0 + 128, 0, 1);
      __builtin_prefetch(a1r + k0 + 128, 0, 1);
    }
    v16bf a0 = frag(a0r + ka, a0r + ka + 8);
    v16bf a1 = frag(a1r + ka, a1r + ka + 8);
    v16bf b0 = frag(b0r + kb, b0r + kb + 8);
    v16bf b1 = frag(b1r + kb, b1r + kb + 8);
    v16bf b2 = frag(b2r + kb, b2r + kb + 8);
    v16bf b3 = frag(b3r + kb, b3r + kb + 8);
    acc[0] = WMMA_BF16(a0, b0, acc[0]);
    acc[1] = WMMA_BF16(a0, b1, acc[1]);
    acc[2] = WMMA_BF16(a0, b2, acc[2]);
    acc[3] = WMMA_BF16(a0, b3, acc[3]);
    acc[4] = WMMA_BF16(a1, b0, acc[4]);
    acc[5] = WMMA_BF16(a1, b1, acc[5]);
    acc[6] = WMMA_BF16(a1, b2, acc[6]);
    acc[7] = WMMA_BF16(a1, b3, acc[7]);
  }
}

// ---- K0: f32 -> bf16 bulk conversion ---------------------------------------
__global__ void k_cvt(const float* __restrict__ s, __bf16* __restrict__ d, int n) {
  int t = blockIdx.x * blockDim.x + threadIdx.x;
  if (t < n) d[t] = f2bf(s[t]);
}

// ---- K1: fused QKV GEMM: Y(4096x1536) = h(4096x512) @ [Wq;Wkv]^T -----------
// q,k stored row-major bf16 (r,512); v stored transposed vT[(b,n,d)][l] bf16.
__global__ void k_qkv(const __bf16* __restrict__ hb, const __bf16* __restrict__ Wqb,
                      const __bf16* __restrict__ Wkvb,
                      __bf16* __restrict__ q, __bf16* __restrict__ k,
                      __bf16* __restrict__ vT) {
  int wave = (blockIdx.x * blockDim.x + threadIdx.x) >> 5;
  int lane = threadIdx.x & 31, half = lane >> 4, l16 = lane & 15;
  int tm = wave / 24, tn = wave % 24;       // 128 x 24 tiles of 32x64
  int m0 = tm * 32, c0 = tn * 64;
  const __bf16* a0r = hb + (size_t)(m0 + l16) * 512;
  const __bf16* a1r = a0r + (size_t)16 * 512;
  const __bf16* br[4];
#pragma unroll
  for (int jb = 0; jb < 4; jb++) {
    int cj = c0 + jb * 16 + l16;
    br[jb] = (cj < 512) ? (Wqb + (size_t)cj * 512)
                        : (Wkvb + (size_t)(cj - 512) * 512);
  }
  v8f acc[8] = {};
  wave_gemm_2x4(a0r, a1r, br[0], br[1], br[2], br[3], 512, half, acc);
#pragma unroll
  for (int ia = 0; ia < 2; ia++) {
#pragma unroll
    for (int jb = 0; jb < 4; jb++) {
      int col = c0 + jb * 16 + l16;
#pragma unroll
      for (int j = 0; j < 8; j++) {
        int r = m0 + ia * 16 + j + half * 8;
        __bf16 val = f2bf(acc[ia * 4 + jb][j]);
        if (col < 512) {
          q[(size_t)r * 512 + col] = val;
        } else if (col < 1024) {
          k[(size_t)r * 512 + (col - 512)] = val;
        } else {
          int c2 = col - 1024, n = c2 >> 6, d = c2 & 63;
          int l = r >> 1, b = r & 1;
          vT[((size_t)((b << 3) + n) * 64 + d) * 2048 + l] = val;
        }
      }
    }
  }
}

// ---- K2: sq[rn] = 0.5*|xs|^2 = (1/16)*|q_head|^2 ---------------------------
__global__ void k_sq(const __bf16* __restrict__ q, const __bf16* __restrict__ k,
                     float* __restrict__ sqq, float* __restrict__ sqk) {
  int t = blockIdx.x * blockDim.x + threadIdx.x;   // 0..32767
  const __bf16* a = q + (size_t)t * 64;
  const __bf16* c = k + (size_t)t * 64;
  float s = 0.f, s2 = 0.f;
#pragma unroll 8
  for (int i = 0; i < 64; i++) {
    float va = bf2f(a[i]); s  += va * va;
    float vc = bf2f(c[i]); s2 += vc * vc;
  }
  sqq[t] = 0.0625f * s;
  sqk[t] = 0.0625f * s2;
}

// ---- K3: projT[m][d] = proj[d][m] * DH^-0.25, stored bf16 ------------------
__global__ void k_projT(const float* __restrict__ proj, __bf16* __restrict__ projT) {
  int t = blockIdx.x * blockDim.x + threadIdx.x;   // 0..16383, t = m*64+d
  int m = t >> 6, d = t & 63;
  projT[t] = f2bf(proj[(size_t)d * 256 + m] * 0.35355339059327373f);  // 64^-0.25
}

// ---- K4: FAVOR+ feature map: exp(xs@proj - sq)/sqrt(M) ---------------------
// MODE 0: q -> qpG[(b,n,l)][m] row-major; MODE 1: k -> kpT[(b,n,m)][l].
template <int MODE>
__global__ void k_prime(const __bf16* __restrict__ src, const float* __restrict__ sq,
                        const __bf16* __restrict__ projT,
                        __bf16* __restrict__ dstQ, __bf16* __restrict__ dstK) {
  int wave = (blockIdx.x * blockDim.x + threadIdx.x) >> 5;
  int lane = threadIdx.x & 31, half = lane >> 4, l16 = lane & 15;
  int tm = wave >> 2, tn = wave & 3;        // 1024 x 4 tiles of 32x64
  int m0 = tm * 32, n0 = tn * 64;
  const __bf16* a0r = src + (size_t)(m0 + l16) * 64;
  const __bf16* a1r = a0r + (size_t)16 * 64;
  const __bf16* b0r = projT + (size_t)(n0 + l16) * 64;
  v8f acc[8] = {};
  wave_gemm_2x4(a0r, a1r, b0r, b0r + 16 * 64, b0r + 32 * 64, b0r + 48 * 64,
                64, half, acc);
#pragma unroll
  for (int ia = 0; ia < 2; ia++) {
#pragma unroll
    for (int jb = 0; jb < 4; jb++) {
      int col = n0 + jb * 16 + l16;
#pragma unroll
      for (int j = 0; j < 8; j++) {
        int rn = m0 + ia * 16 + j + half * 8;
        float e = expf(acc[ia * 4 + jb][j] - sq[rn]) * 0.0625f;   // * M^-0.5
        int r = rn >> 3, n = rn & 7;
        int l = r >> 1, b = r & 1;
        int bn = (b << 3) + n;
        if (MODE == 0) dstQ[((size_t)bn * 2048 + l) * 256 + col] = f2bf(e);
        else           dstK[((size_t)bn * 256 + col) * 2048 + l] = f2bf(e);
      }
    }
  }
}

// ---- K5: z[bn][m] = sum_l kp ------------------------------------------------
__global__ void k_z(const __bf16* __restrict__ kpT, float* __restrict__ z) {
  int wave = (blockIdx.x * blockDim.x + threadIdx.x) >> 5;   // 0..4095
  int lane = threadIdx.x & 31;
  const __bf16* row = kpT + (size_t)wave * 2048;
  float s = 0.f;
  for (int i = lane; i < 2048; i += 32) s += bf2f(row[i]);
#pragma unroll
  for (int off = 16; off >= 1; off >>= 1) s += __shfl_xor(s, off);
  if (lane == 0) z[wave] = s;
}

// ---- K6: S[bn] = Kp^T(256x2048) @ V(2048x64); stored transposed ST[(bn,d)][m]
__global__ void k_S(const __bf16* __restrict__ kpT, const __bf16* __restrict__ vT,
                    __bf16* __restrict__ ST) {
  int wave = (blockIdx.x * blockDim.x + threadIdx.x) >> 5;   // 0..127
  int lane = threadIdx.x & 31, half = lane >> 4, l16 = lane & 15;
  int bn = wave >> 3, mt = wave & 7;
  int m0 = mt * 32;
  const __bf16* a0r = kpT + ((size_t)bn * 256 + m0 + l16) * 2048;
  const __bf16* a1r = a0r + (size_t)16 * 2048;
  const __bf16* b0r = vT + ((size_t)bn * 64 + l16) * 2048;
  v8f acc[8] = {};
  wave_gemm_2x4(a0r, a1r, b0r, b0r + (size_t)16 * 2048, b0r + (size_t)32 * 2048,
                b0r + (size_t)48 * 2048, 2048, half, acc);
#pragma unroll
  for (int ia = 0; ia < 2; ia++) {
#pragma unroll
    for (int jb = 0; jb < 4; jb++) {
#pragma unroll
      for (int j = 0; j < 8; j++) {
        int m = m0 + ia * 16 + j + half * 8, d = jb * 16 + l16;
        ST[((size_t)bn * 64 + d) * 256 + m] = f2bf(acc[ia * 4 + jb][j]);
      }
    }
  }
}

// ---- K7: denom[bn][l] = scale * qp . z + EPS -------------------------------
__global__ void k_denom(const __bf16* __restrict__ qpG, const float* __restrict__ z,
                        float* __restrict__ den) {
  int wave = (blockIdx.x * blockDim.x + threadIdx.x) >> 5;   // 0..32767
  int lane = threadIdx.x & 31;
  int bn = wave >> 11;
  const __bf16* row = qpG + (size_t)wave * 256;
  const float*  zr  = z + (size_t)bn * 256;
  float s = 0.f;
  for (int i = lane; i < 256; i += 32) s += bf2f(row[i]) * zr[i];
#pragma unroll
  for (int off = 16; off >= 1; off >>= 1) s += __shfl_xor(s, off);
  if (lane == 0) den[wave] = 0.125f * s + 1e-5f;
}

// ---- K8: vec = scale * (Qp @ S) / denom ------------------------------------
__global__ void k_vec(const __bf16* __restrict__ qpG, const __bf16* __restrict__ ST,
                      const float* __restrict__ den, __bf16* __restrict__ vec) {
  int wave = (blockIdx.x * blockDim.x + threadIdx.x) >> 5;   // 0..1023
  int lane = threadIdx.x & 31, half = lane >> 4, l16 = lane & 15;
  int bn = wave >> 6, lt = wave & 63;
  int m0 = lt * 32;
  const __bf16* a0r = qpG + ((size_t)bn * 2048 + m0 + l16) * 256;
  const __bf16* a1r = a0r + (size_t)16 * 256;
  const __bf16* b0r = ST + ((size_t)bn * 64 + l16) * 256;
  v8f acc[8] = {};
  wave_gemm_2x4(a0r, a1r, b0r, b0r + 16 * 256, b0r + 32 * 256, b0r + 48 * 256,
                256, half, acc);
  int n = bn & 7, b = bn >> 3;
#pragma unroll
  for (int ia = 0; ia < 2; ia++) {
#pragma unroll
    for (int jb = 0; jb < 4; jb++) {
#pragma unroll
      for (int j = 0; j < 8; j++) {
        int l = m0 + ia * 16 + j + half * 8, d = jb * 16 + l16;
        float val = 0.125f * acc[ia * 4 + jb][j] / den[(size_t)bn * 2048 + l];
        vec[(size_t)(l * 2 + b) * 512 + n * 64 + d] = f2bf(val);
      }
    }
  }
}

// ---- K9: x = h + vec @ Wo^T ------------------------------------------------
__global__ void k_out(const __bf16* __restrict__ vec, const __bf16* __restrict__ Wob,
                      const float* __restrict__ h, float* __restrict__ x) {
  int wave = (blockIdx.x * blockDim.x + threadIdx.x) >> 5;   // 0..1023
  int lane = threadIdx.x & 31, half = lane >> 4, l16 = lane & 15;
  int tm = wave >> 3, tn = wave & 7;        // 128 x 8 tiles of 32x64
  int m0 = tm * 32, c0 = tn * 64;
  const __bf16* a0r = vec + (size_t)(m0 + l16) * 512;
  const __bf16* a1r = a0r + (size_t)16 * 512;
  const __bf16* b0r = Wob + (size_t)(c0 + l16) * 512;
  v8f acc[8] = {};
  wave_gemm_2x4(a0r, a1r, b0r, b0r + 16 * 512, b0r + 32 * 512, b0r + 48 * 512,
                512, half, acc);
#pragma unroll
  for (int ia = 0; ia < 2; ia++) {
#pragma unroll
    for (int jb = 0; jb < 4; jb++) {
      int c = c0 + jb * 16 + l16;
#pragma unroll
      for (int j = 0; j < 8; j++) {
        int r = m0 + ia * 16 + j + half * 8;
        x[(size_t)r * 512 + c] = h[(size_t)r * 512 + c] + acc[ia * 4 + jb][j];
      }
    }
  }
}

// ---- K10: LayerNorm per row -------------------------------------------------
__global__ void k_ln(const float* __restrict__ x, const float* __restrict__ w,
                     const float* __restrict__ b, float* __restrict__ out) {
  int wave = (blockIdx.x * blockDim.x + threadIdx.x) >> 5;   // 0..4095 = row
  int lane = threadIdx.x & 31;
  const float* row = x + (size_t)wave * 512;
  float xv[16]; float s = 0.f, s2 = 0.f;
#pragma unroll
  for (int i = 0; i < 16; i++) {
    float v = row[lane + i * 32];
    xv[i] = v; s += v; s2 += v * v;
  }
#pragma unroll
  for (int off = 16; off >= 1; off >>= 1) {
    s += __shfl_xor(s, off); s2 += __shfl_xor(s2, off);
  }
  float mu  = s * (1.0f / 512.0f);
  float var = s2 * (1.0f / 512.0f) - mu * mu;
  float rs  = rsqrtf(var + 1e-5f);
  float* orow = out + (size_t)wave * 512;
#pragma unroll
  for (int i = 0; i < 16; i++) {
    int c = lane + i * 32;
    orow[c] = w[c] * (xv[i] - mu) * rs + b[c];
  }
}

extern "C" void kernel_launch(void* const* d_in, const int* in_sizes, int n_in,
                              void* d_out, int out_size, void* d_ws, size_t ws_size,
                              hipStream_t stream) {
  (void)in_sizes; (void)n_in; (void)out_size; (void)ws_size;
  const float* h    = (const float*)d_in[0];
  const float* Wq   = (const float*)d_in[1];
  const float* Wkv  = (const float*)d_in[2];
  const float* Wo   = (const float*)d_in[3];
  const float* ln_w = (const float*)d_in[4];
  const float* ln_b = (const float*)d_in[5];
  const float* proj = (const float*)d_in[6];

  char* p = (char*)d_ws;
  auto carve = [&](size_t bytes) -> char* {
    char* r = p; p += (bytes + 255) & ~(size_t)255; return r;
  };
  __bf16* hb   = (__bf16*)carve((size_t)N_ * DM_ * 2);             // 4 MB
  __bf16* Wqb  = (__bf16*)carve((size_t)512 * 512 * 2);
  __bf16* Wkvb = (__bf16*)carve((size_t)1024 * 512 * 2);
  __bf16* Wob  = (__bf16*)carve((size_t)512 * 512 * 2);
  __bf16* q    = (__bf16*)carve((size_t)RN_ * DH_ * 2);            // 4 MB
  __bf16* k    = (__bf16*)carve((size_t)RN_ * DH_ * 2);            // 4 MB
  __bf16* vT   = (__bf16*)carve((size_t)B_ * NH_ * DH_ * L_ * 2);  // 4 MB
  __bf16* qpG  = (__bf16*)carve((size_t)B_ * NH_ * L_ * M_ * 2);   // 16 MB
  __bf16* kpT  = (__bf16*)carve((size_t)B_ * NH_ * M_ * L_ * 2);   // 16 MB
  __bf16* ST   = (__bf16*)carve((size_t)B_ * NH_ * DH_ * M_ * 2);  // 0.5 MB
  __bf16* vec  = (__bf16*)carve((size_t)N_ * DM_ * 2);             // 4 MB
  __bf16* projT= (__bf16*)carve((size_t)M_ * DH_ * 2);
  float* sqq   = (float*)carve((size_t)RN_ * 4);
  float* sqk   = (float*)carve((size_t)RN_ * 4);
  float* z     = (float*)carve((size_t)B_ * NH_ * M_ * 4);
  float* den   = (float*)carve((size_t)B_ * NH_ * L_ * 4);
  float* x     = (float*)carve((size_t)N_ * DM_ * 4);              // 8 MB

  dim3 blk(256);
  k_cvt   <<<8192, blk, 0, stream>>>(h,   hb,   N_ * DM_);
  k_cvt   <<<1024, blk, 0, stream>>>(Wq,  Wqb,  512 * 512);
  k_cvt   <<<2048, blk, 0, stream>>>(Wkv, Wkvb, 1024 * 512);
  k_cvt   <<<1024, blk, 0, stream>>>(Wo,  Wob,  512 * 512);
  k_projT <<<64,   blk, 0, stream>>>(proj, projT);
  k_qkv   <<<384,  blk, 0, stream>>>(hb, Wqb, Wkvb, q, k, vT);
  k_sq    <<<128,  blk, 0, stream>>>(q, k, sqq, sqk);
  k_prime<0><<<512, blk, 0, stream>>>(q, sqq, projT, qpG, kpT);
  k_prime<1><<<512, blk, 0, stream>>>(k, sqk, projT, qpG, kpT);
  k_z     <<<512,  blk, 0, stream>>>(kpT, z);
  k_S     <<<16,   blk, 0, stream>>>(kpT, vT, ST);
  k_denom <<<4096, blk, 0, stream>>>(qpG, z, den);
  k_vec   <<<128,  blk, 0, stream>>>(qpG, ST, den, vec);
  k_out   <<<128,  blk, 0, stream>>>(vec, Wob, h, x);
  k_ln    <<<512,  blk, 0, stream>>>(x, ln_w, ln_b, (float*)d_out);
}
